// SAGEClassifier_85564338471312
// MI455X (gfx1250) — matrix operand order repaired and checked by reading, not compiled, not run on hardware
//
#include <hip/hip_runtime.h>
#include <hip/hip_bf16.h>

#define N_NODES 100000
#define N_EDGES 1600000
#define IN_DIM  128
#define HID     128
#define INT_DIM 512
#define OUT_DIM 3

typedef __bf16 bf16_t;
typedef __attribute__((ext_vector_type(16))) __bf16 v16bf;
typedef __attribute__((ext_vector_type(8)))  __bf16 bf16x8;
typedef __attribute__((ext_vector_type(8)))  float  v8f;
typedef __attribute__((ext_vector_type(4)))  unsigned int u32x4;
typedef __attribute__((ext_vector_type(8)))  int i32x8;
typedef __attribute__((ext_vector_type(4)))  int i32x4;

// ---------------- elementwise helpers ----------------

__global__ void k_zero_f32(float* __restrict__ p, int n) {
    int i = blockIdx.x * blockDim.x + threadIdx.x;
    if (i < n) p[i] = 0.0f;
}

__global__ void k_f32_to_bf16(const float* __restrict__ in, bf16_t* __restrict__ out, int n) {
    int i = blockIdx.x * blockDim.x + threadIdx.x;
    if (i < n) out[i] = (bf16_t)in[i];
}

// Wb[o][0:Kh] = Wl[o][:], Wb[o][Kh:2Kh] = Wr[o][:]   (bf16 packed pair)
__global__ void k_pack_pair(const float* __restrict__ Wl, const float* __restrict__ Wr,
                            bf16_t* __restrict__ Wb, int O, int Kh) {
    int i = blockIdx.x * blockDim.x + threadIdx.x;
    int n = O * 2 * Kh;
    if (i >= n) return;
    int o = i / (2 * Kh);
    int d = i - o * 2 * Kh;
    float v = (d < Kh) ? Wl[o * Kh + d] : Wr[o * Kh + (d - Kh)];
    Wb[i] = (bf16_t)v;
}

// ---------------- graph aggregation ----------------

__global__ void k_count(const int* __restrict__ ei, float* __restrict__ cnt) {
    int e = blockIdx.x * blockDim.x + threadIdx.x;
    if (e < N_EDGES) atomicAdd(&cnt[ei[N_EDGES + e]], 1.0f);
}

// one thread per (edge, dim); D fixed at 128
__global__ void k_scatter(const bf16_t* __restrict__ feat, const int* __restrict__ ei,
                          float* __restrict__ agg) {
    int id = blockIdx.x * blockDim.x + threadIdx.x;
    int e = id >> 7;
    int d = id & 127;
    if (e < N_EDGES) {
        int s = ei[e];
        int t = ei[N_EDGES + e];
        atomicAdd(&agg[t * 128 + d], (float)feat[s * 128 + d]);
    }
}

// A'[i][0:128] = agg[i]/max(cnt,1) (bf16),  A'[i][128:256] = hprev[i] (bf16 copy)
__global__ void k_prep(const float* __restrict__ agg, const float* __restrict__ cnt,
                       const bf16_t* __restrict__ hprev, bf16_t* __restrict__ Ab) {
    int id = blockIdx.x * blockDim.x + threadIdx.x;
    if (id >= N_NODES * 256) return;
    int i = id >> 8;
    int d = id & 255;
    if (d < 128) {
        float c = cnt[i];
        Ab[id] = (bf16_t)(agg[i * 128 + d] / fmaxf(c, 1.0f));
    } else {
        Ab[id] = hprev[i * 128 + (d - 128)];
    }
}

// ---------------- WMMA GEMM: out[N, 128*CT] = A[N,K] @ W[O,K]^T + bias ----------------
// 256 threads = 8 wave32s; wave owns CT 16x16 tiles -> workgroup owns a full 16-row block.
// A-tile (16 x K bf16) staged into LDS once per workgroup by the Tensor Data Mover with
// 4-DWORD row padding; waves read fragments via ds_load_b128. B streams from global (L2).
// MODE 0: L2-normalize rows then ReLU.  MODE 1: ReLU only.
template <int K, int CT, int MODE>
__global__ __launch_bounds__(256) void k_gemm_wmma(
    const bf16_t* __restrict__ A, const bf16_t* __restrict__ W,
    const float* __restrict__ bias, bf16_t* __restrict__ out)
{
    constexpr int O = 128 * CT;
    constexpr int ROWB = 2 * K + 16;          // padded LDS row stride (bytes)
    __shared__ __align__(16) unsigned char shA[16 * ROWB];

    const int lane = threadIdx.x & 31;
    const int wave = threadIdx.x >> 5;
    const int rowbase = blockIdx.x * 16;      // N_NODES % 16 == 0
    const int m16 = lane & 15;
    const int kh  = lane >> 4;

    // ---- TDM: DMA the 16 x K bf16 A-tile into LDS (wave 0 issues; EXEC ignored) ----
    if (wave == 0) {
        unsigned lds_off = (unsigned)(uintptr_t)(&shA[0]);
        unsigned long long ga =
            (unsigned long long)(uintptr_t)(A + (size_t)rowbase * K);
        u32x4 g0;
        g0[0] = 1u;                                        // count=1 (valid descriptor)
        g0[1] = lds_off;                                   // lds_addr (bytes)
        g0[2] = (unsigned)(ga & 0xFFFFFFFFu);              // global_addr[31:0]
        g0[3] = (unsigned)((ga >> 32) & 0x01FFFFFFu)       // global_addr[56:32]
              | 0x80000000u;                               // type=2 ("image")
        constexpr unsigned padc = 3u;                      // pad_amount: 4 DWORDs
        constexpr unsigned intc = (K == 256) ? 6u : 7u;    // pad_interval: K/2 DWORDs/row
        i32x8 g1;
        g1[0] = (int)((1u << 16)                           // data_size = 2 bytes
                    | (1u << 20)                           // pad_enable
                    | (intc << 22) | (padc << 25));
        g1[1] = (int)(((unsigned)K & 0xFFFFu) << 16);      // tensor_dim0[15:0]
        g1[2] = (int)(((unsigned)K >> 16) | (16u << 16));  // tensor_dim0[31:16] | tensor_dim1.lo=16
        g1[3] = (int)((unsigned)K << 16);                  // tensor_dim1.hi=0 | tile_dim0=K
        g1[4] = 16;                                        // tile_dim1 = 16 rows
        g1[5] = K;                                         // tensor_dim0_stride (elements)
        g1[6] = 0;
        g1[7] = 0;
        i32x4 g2; g2[0] = 0; g2[1] = 0; g2[2] = 0; g2[3] = 0;
        i32x4 g3 = g2;
        i32x8 g4;
        #pragma unroll
        for (int j = 0; j < 8; ++j) g4[j] = 0;
        __builtin_amdgcn_tensor_load_to_lds(g0, g1, g2, g3, g4, 0);
        __builtin_amdgcn_s_wait_tensorcnt(0);
    }
    __syncthreads();

    v8f acc[CT];
    #pragma unroll
    for (int ct = 0; ct < CT; ++ct)
        #pragma unroll
        for (int r = 0; r < 8; ++r) acc[ct][r] = 0.0f;

    const bf16_t* arow = (const bf16_t*)(shA + m16 * ROWB);

    #pragma unroll 2
    for (int k = 0; k < K; k += 32) {
        // A fragment 16x32 bf16 from LDS: lane(m16,kh) holds K = k+kh*16 .. +15
        bf16x8 alo = *(const bf16x8*)(arow + k + kh * 16);
        bf16x8 ahi = *(const bf16x8*)(arow + k + kh * 16 + 8);
        v16bf av;
        #pragma unroll
        for (int j = 0; j < 8; ++j) { av[j] = alo[j]; av[j + 8] = ahi[j]; }

        #pragma unroll
        for (int ct = 0; ct < CT; ++ct) {
            int col = (wave * CT + ct) * 16 + m16;
            // B fragment 32x16: lane(n=m16,kh) holds K = k+kh*16 .. +15 of column n
            v16bf bv = *(const v16bf*)(W + (size_t)col * K + k + kh * 16);
            acc[ct] = __builtin_amdgcn_wmma_f32_16x16x32_bf16(
                false, av, false, bv, (short)0, acc[ct], false, false);
        }
    }

    // C layout: lane(n=m16, mh) vgpr r -> row = rowbase + mh*8 + r, col = tile*16 + n
    const int mh = lane >> 4;
    float vals[CT][8];
    #pragma unroll
    for (int ct = 0; ct < CT; ++ct) {
        float b = bias[(wave * CT + ct) * 16 + m16];
        #pragma unroll
        for (int r = 0; r < 8; ++r) vals[ct][r] = acc[ct][r] + b;
    }

    if (MODE == 0) {
        __shared__ float rowsq[16];
        if (threadIdx.x < 16) rowsq[threadIdx.x] = 0.0f;
        __syncthreads();
        float psum[8];
        #pragma unroll
        for (int r = 0; r < 8; ++r) {
            psum[r] = 0.0f;
            #pragma unroll
            for (int ct = 0; ct < CT; ++ct) psum[r] += vals[ct][r] * vals[ct][r];
        }
        #pragma unroll
        for (int r = 0; r < 8; ++r) atomicAdd(&rowsq[mh * 8 + r], psum[r]);
        __syncthreads();
        float sc[8];
        #pragma unroll
        for (int r = 0; r < 8; ++r)
            sc[r] = 1.0f / fmaxf(sqrtf(rowsq[mh * 8 + r]), 1e-12f);
        #pragma unroll
        for (int ct = 0; ct < CT; ++ct)
            #pragma unroll
            for (int r = 0; r < 8; ++r)
                vals[ct][r] = fmaxf(vals[ct][r] * sc[r], 0.0f);
    } else {
        #pragma unroll
        for (int ct = 0; ct < CT; ++ct)
            #pragma unroll
            for (int r = 0; r < 8; ++r)
                vals[ct][r] = fmaxf(vals[ct][r], 0.0f);
    }

    #pragma unroll
    for (int ct = 0; ct < CT; ++ct) {
        int col = (wave * CT + ct) * 16 + m16;
        #pragma unroll
        for (int r = 0; r < 8; ++r)
            out[(size_t)(rowbase + mh * 8 + r) * O + col] = (bf16_t)vals[ct][r];
    }
}

// ---------------- final 512 -> 3 head (fp32 out) ----------------

__global__ void k_final(const bf16_t* __restrict__ h, const float* __restrict__ Wc,
                        const float* __restrict__ bc, float* __restrict__ out) {
    int i = blockIdx.x * blockDim.x + threadIdx.x;
    if (i >= N_NODES) return;
    float a0 = bc[0], a1 = bc[1], a2 = bc[2];
    const bf16_t* hp = h + (size_t)i * INT_DIM;
    for (int k = 0; k < INT_DIM; ++k) {
        float v = (float)hp[k];
        a0 += v * Wc[k];
        a1 += v * Wc[INT_DIM + k];
        a2 += v * Wc[2 * INT_DIM + k];
    }
    out[i * 3 + 0] = a0;
    out[i * 3 + 1] = a1;
    out[i * 3 + 2] = a2;
}

// ---------------- launcher ----------------

extern "C" void kernel_launch(void* const* d_in, const int* in_sizes, int n_in,
                              void* d_out, int out_size, void* d_ws, size_t ws_size,
                              hipStream_t stream) {
    (void)in_sizes; (void)n_in; (void)out_size; (void)ws_size;

    const float* x   = (const float*)d_in[0];
    const int*   ei  = (const int*)d_in[1];
    const float* W1l = (const float*)d_in[2];
    const float* b1l = (const float*)d_in[3];
    const float* W1r = (const float*)d_in[4];
    const float* W2l = (const float*)d_in[5];
    const float* b2l = (const float*)d_in[6];
    const float* W2r = (const float*)d_in[7];
    const float* W3l = (const float*)d_in[8];
    const float* b3l = (const float*)d_in[9];
    const float* W3r = (const float*)d_in[10];
    const float* Wi  = (const float*)d_in[11];
    const float* bi  = (const float*)d_in[12];
    const float* Wc  = (const float*)d_in[13];
    const float* bc  = (const float*)d_in[14];
    float* out = (float*)d_out;

    char* ws = (char*)d_ws;
    size_t off = 0;
    auto alloc = [&](size_t bytes) -> char* {
        char* p = ws + off;
        off = (off + bytes + 255) & ~(size_t)255;
        return p;
    };
    float*  agg = (float*) alloc((size_t)N_NODES * 128 * 4);
    float*  cnt = (float*) alloc((size_t)N_NODES * 4);
    bf16_t* Ab  = (bf16_t*)alloc((size_t)N_NODES * 256 * 2);
    bf16_t* xb  = (bf16_t*)alloc((size_t)N_NODES * 128 * 2);
    bf16_t* hb1 = (bf16_t*)alloc((size_t)N_NODES * 128 * 2);
    bf16_t* hb2 = (bf16_t*)alloc((size_t)N_NODES * 128 * 2);
    bf16_t* hb3 = (bf16_t*)alloc((size_t)N_NODES * 512 * 2);
    bf16_t* hb4 = (bf16_t*)alloc((size_t)N_NODES * 512 * 2);
    bf16_t* Wb1 = (bf16_t*)alloc(128 * 256 * 2);
    bf16_t* Wb2 = (bf16_t*)alloc(128 * 256 * 2);
    bf16_t* Wb3 = (bf16_t*)alloc(512 * 256 * 2);
    bf16_t* Wib = (bf16_t*)alloc(512 * 512 * 2);

    const int TB = 256;
    #define GRID(n) ((((int)(n)) + TB - 1) / TB)

    // weight prep (bf16)
    k_pack_pair<<<GRID(128 * 256), TB, 0, stream>>>(W1l, W1r, Wb1, 128, 128);
    k_pack_pair<<<GRID(128 * 256), TB, 0, stream>>>(W2l, W2r, Wb2, 128, 128);
    k_pack_pair<<<GRID(512 * 256), TB, 0, stream>>>(W3l, W3r, Wb3, 512, 128);
    k_f32_to_bf16<<<GRID(512 * 512), TB, 0, stream>>>(Wi, Wib, 512 * 512);
    k_f32_to_bf16<<<GRID(N_NODES * 128), TB, 0, stream>>>(x, xb, N_NODES * 128);

    // degree counts (shared by all layers)
    k_zero_f32<<<GRID(N_NODES), TB, 0, stream>>>(cnt, N_NODES);
    k_count<<<GRID(N_EDGES), TB, 0, stream>>>(ei, cnt);

    // ---- SAGE layer 1: x -> hb1 (128) ----
    k_zero_f32<<<GRID(N_NODES * 128), TB, 0, stream>>>(agg, N_NODES * 128);
    k_scatter<<<GRID((size_t)N_EDGES * 128), TB, 0, stream>>>(xb, ei, agg);
    k_prep<<<GRID(N_NODES * 256), TB, 0, stream>>>(agg, cnt, xb, Ab);
    k_gemm_wmma<256, 1, 0><<<N_NODES / 16, 256, 0, stream>>>(Ab, Wb1, b1l, hb1);

    // ---- SAGE layer 2: hb1 -> hb2 (128) ----
    k_zero_f32<<<GRID(N_NODES * 128), TB, 0, stream>>>(agg, N_NODES * 128);
    k_scatter<<<GRID((size_t)N_EDGES * 128), TB, 0, stream>>>(hb1, ei, agg);
    k_prep<<<GRID(N_NODES * 256), TB, 0, stream>>>(agg, cnt, hb1, Ab);
    k_gemm_wmma<256, 1, 0><<<N_NODES / 16, 256, 0, stream>>>(Ab, Wb2, b2l, hb2);

    // ---- SAGE layer 3: hb2 -> hb3 (512) ----
    k_zero_f32<<<GRID(N_NODES * 128), TB, 0, stream>>>(agg, N_NODES * 128);
    k_scatter<<<GRID((size_t)N_EDGES * 128), TB, 0, stream>>>(hb2, ei, agg);
    k_prep<<<GRID(N_NODES * 256), TB, 0, stream>>>(agg, cnt, hb2, Ab);
    k_gemm_wmma<256, 4, 0><<<N_NODES / 16, 256, 0, stream>>>(Ab, Wb3, b3l, hb3);

    // ---- dense 512 -> 512 + ReLU ----
    k_gemm_wmma<512, 4, 1><<<N_NODES / 16, 256, 0, stream>>>(hb3, Wib, bi, hb4);

    // ---- head 512 -> 3 ----
    k_final<<<GRID(N_NODES), TB, 0, stream>>>(hb4, Wc, bc, out);
    #undef GRID
}
